// CuPyLinear_17403207483562
// MI455X (gfx1250) — matrix operand order repaired and checked by reading, not compiled, not run on hardware
//
#include <hip/hip_runtime.h>

// Problem constants (fixed by the reference implementation).
#define OUT_F 2048
#define IN_F  2048
#define N_TOK 512

// GEMM tiling.
#define K_SLAB  64                 // K per LDS slab
#define NSLAB   (IN_F / K_SLAB)    // 32
#define LDS_STRIDE 72              // 64 halfs + 8 pad (144 B: 16B-aligned, bank-spread)
#define A_HALFS (128 * LDS_STRIDE) // 9216
#define B_HALFS (64  * LDS_STRIDE) // 4608
#define BUF_HALFS (A_HALFS + B_HALFS)

typedef __attribute__((ext_vector_type(16))) _Float16 v16h;
typedef __attribute__((ext_vector_type(8)))  _Float16 v8h;
typedef __attribute__((ext_vector_type(8)))  float    v8f;
typedef __attribute__((ext_vector_type(4)))  int      v4i;

#if defined(__HIP_DEVICE_COMPILE__) && __has_builtin(__builtin_amdgcn_global_load_async_to_lds_b128)
#define USE_ASYNC_LDS 1
#else
#define USE_ASYNC_LDS 0
#endif

// ---------------------------------------------------------------------------
// Phase 1a: zero the dense f16 weight buffer (8 MB) with 16B stores.
// ---------------------------------------------------------------------------
__global__ void zero_wd(uint4* __restrict__ wd, long n16) {
    long i = (long)blockIdx.x * blockDim.x + threadIdx.x;
    long stride = (long)gridDim.x * blockDim.x;
    uint4 z = {0u, 0u, 0u, 0u};
    for (; i < n16; i += stride) wd[i] = z;
}

// ---------------------------------------------------------------------------
// Phase 1b: convert x (fp32) -> xh (f16).
// ---------------------------------------------------------------------------
__global__ void cvt_x(const float* __restrict__ x, _Float16* __restrict__ xh, long n) {
    long i = (long)blockIdx.x * blockDim.x + threadIdx.x;
    long stride = (long)gridDim.x * blockDim.x;
    for (; i < n; i += stride) xh[i] = (_Float16)x[i];
}

// ---------------------------------------------------------------------------
// Phase 2: scatter CSR row r into dense row wd[r][*] (f16), folding duplicate
// sorted column indices (they sum, per reference). One thread per row.
// ---------------------------------------------------------------------------
__global__ void scatter_csr(const float* __restrict__ data,
                            const int*   __restrict__ indices,
                            const int*   __restrict__ indptr,
                            _Float16*    __restrict__ wd) {
    int r = blockIdx.x * blockDim.x + threadIdx.x;
    if (r >= OUT_F) return;
    int s = indptr[r], e = indptr[r + 1];
    _Float16* row = wd + (long)r * IN_F;
    int j = s;
    while (j < e) {
        int   c   = indices[j];
        float acc = data[j];
        ++j;
        while (j < e && indices[j] == c) { acc += data[j]; ++j; }
        row[c] = (_Float16)acc;
    }
}

// ---------------------------------------------------------------------------
// Copy helpers for the GEMM pipeline.
// Slab geometry: A = 128 rows x 64 halfs (8 x 16B segs/row -> 1024 segs),
//                B =  64 rows x 64 halfs (512 segs). 256 threads.
// ---------------------------------------------------------------------------
__device__ __forceinline__ void async_cp16(const _Float16* g, _Float16* l) {
#if USE_ASYNC_LDS
    // Probe-derived signature: (v4i addrspace(1)*, v4i addrspace(3)*, imm, imm)
    typedef v4i __attribute__((address_space(1)))* gp_t;
    typedef v4i __attribute__((address_space(3)))* sp_t;
    __builtin_amdgcn_global_load_async_to_lds_b128(
        (gp_t)(unsigned long long)(uintptr_t)g,
        (sp_t)(unsigned int)(uintptr_t)l, 0, 0);
#else
    (void)g; (void)l;
#endif
}

__device__ __forceinline__ void wait_async0() {
#if USE_ASYNC_LDS
#if __has_builtin(__builtin_amdgcn_s_wait_asynccnt)
    __builtin_amdgcn_s_wait_asynccnt(0);
#else
    asm volatile("s_wait_asynccnt 0" ::: "memory");
#endif
#endif
}

struct SlabRegs { uint4 a[4]; uint4 b[2]; };

__device__ __forceinline__ void gload_slab(const _Float16* __restrict__ xh,
                                           const _Float16* __restrict__ wd,
                                           int mBase, int n0, int k0, int tid,
                                           SlabRegs& r) {
#pragma unroll
    for (int i = 0; i < 4; ++i) {
        int id = tid + i * 256, row = id >> 3, s8 = id & 7;
        r.a[i] = *(const uint4*)(xh + (long)(mBase + row) * IN_F + k0 + s8 * 8);
    }
#pragma unroll
    for (int i = 0; i < 2; ++i) {
        int id = tid + i * 256, row = id >> 3, s8 = id & 7;
        r.b[i] = *(const uint4*)(wd + (long)(n0 + row) * IN_F + k0 + s8 * 8);
    }
}

__device__ __forceinline__ void sstore_slab(_Float16* bufA, _Float16* bufB,
                                            int tid, const SlabRegs& r) {
#pragma unroll
    for (int i = 0; i < 4; ++i) {
        int id = tid + i * 256, row = id >> 3, s8 = id & 7;
        *(uint4*)(bufA + row * LDS_STRIDE + s8 * 8) = r.a[i];
    }
#pragma unroll
    for (int i = 0; i < 2; ++i) {
        int id = tid + i * 256, row = id >> 3, s8 = id & 7;
        *(uint4*)(bufB + row * LDS_STRIDE + s8 * 8) = r.b[i];
    }
}

__device__ __forceinline__ void copy_slab_async(const _Float16* __restrict__ xh,
                                                const _Float16* __restrict__ wd,
                                                int mBase, int n0, int k0, int tid,
                                                _Float16* bufA, _Float16* bufB) {
#pragma unroll
    for (int i = 0; i < 4; ++i) {
        int id = tid + i * 256, row = id >> 3, s8 = id & 7;
        async_cp16(xh + (long)(mBase + row) * IN_F + k0 + s8 * 8,
                   bufA + row * LDS_STRIDE + s8 * 8);
    }
#pragma unroll
    for (int i = 0; i < 2; ++i) {
        int id = tid + i * 256, row = id >> 3, s8 = id & 7;
        async_cp16(wd + (long)(n0 + row) * IN_F + k0 + s8 * 8,
                   bufB + row * LDS_STRIDE + s8 * 8);
    }
}

// ---------------------------------------------------------------------------
// Compute one K-slab out of LDS: 8 WMMAs per wave (4 N-subtiles x 2 k-steps).
// Fragment layouts per ISA 7.12.2 (wave32):
//   A 16x32 f16: lane holds halfs [k+8*lhalf .. +7] and [k+16+8*lhalf .. +7]
//   B 32x16 f16: lane (col = lane&15) holds halfs [k+8*lhalf .. +7] and [k+16+8*lhalf .. +7]
// ---------------------------------------------------------------------------
__device__ __forceinline__ void compute_slab(const _Float16* bufA, const _Float16* bufB,
                                             int wave, int lrow, int lhalf, v8f c[4]) {
    const _Float16* ap = bufA + (wave * 16 + lrow) * LDS_STRIDE + lhalf * 8;
#pragma unroll
    for (int ks = 0; ks < K_SLAB; ks += 32) {
        v8h alo = *(const v8h*)(ap + ks);
        v8h ahi = *(const v8h*)(ap + ks + 16);
        v16h a  = __builtin_shufflevector(alo, ahi,
                      0,1,2,3,4,5,6,7,8,9,10,11,12,13,14,15);
#pragma unroll
        for (int sub = 0; sub < 4; ++sub) {
            const _Float16* bp = bufB + (sub * 16 + lrow) * LDS_STRIDE + ks + lhalf * 8;
            v8h blo = *(const v8h*)(bp);
            v8h bhi = *(const v8h*)(bp + 16);
            v16h b  = __builtin_shufflevector(blo, bhi,
                          0,1,2,3,4,5,6,7,8,9,10,11,12,13,14,15);
            c[sub] = __builtin_amdgcn_wmma_f32_16x16x32_f16(false, a, false, b,
                                                            (short)0, c[sub],
                                                            false, false);
        }
    }
}

// ---------------------------------------------------------------------------
// Phase 3: dense GEMM y[512 x 2048] = xh * wd^T, double-buffered through LDS.
// Block = 256 threads (8 waves), tile 128(M) x 64(N); wave w owns rows
// w*16..w*16+15 (A private per wave, B shared by all 8 waves -> LDS reuse).
// ---------------------------------------------------------------------------
__global__ __launch_bounds__(256)
void spmm_gemm_wmma(const _Float16* __restrict__ xh,
                    const _Float16* __restrict__ wd,
                    float*          __restrict__ y) {
    __shared__ __align__(16) _Float16 smem[2 * BUF_HALFS];

    const int tid   = threadIdx.x;
    const int lane  = tid & 31;
    const int wave  = tid >> 5;
    const int lrow  = lane & 15;
    const int lhalf = lane >> 4;
    const int mBase = blockIdx.x * 128;
    const int n0    = blockIdx.y * 64;

    v8f c[4] = {v8f{}, v8f{}, v8f{}, v8f{}};

#if USE_ASYNC_LDS
    copy_slab_async(xh, wd, mBase, n0, 0, tid, smem, smem + A_HALFS);
    wait_async0();
    __syncthreads();
    for (int s = 0; s < NSLAB; ++s) {
        _Float16* cur = smem + (s & 1) * BUF_HALFS;
        _Float16* nxt = smem + ((s + 1) & 1) * BUF_HALFS;
        if (s + 1 < NSLAB)
            copy_slab_async(xh, wd, mBase, n0, (s + 1) * K_SLAB, tid,
                            nxt, nxt + A_HALFS);
        compute_slab(cur, cur + A_HALFS, wave, lrow, lhalf, c);
        wait_async0();
        __syncthreads();
    }
#else
    SlabRegs r;
    gload_slab(xh, wd, mBase, n0, 0, tid, r);
    sstore_slab(smem, smem + A_HALFS, tid, r);
    __syncthreads();
    for (int s = 0; s < NSLAB; ++s) {
        _Float16* cur = smem + (s & 1) * BUF_HALFS;
        _Float16* nxt = smem + ((s + 1) & 1) * BUF_HALFS;
        bool more = (s + 1 < NSLAB);
        if (more) gload_slab(xh, wd, mBase, n0, (s + 1) * K_SLAB, tid, r);
        compute_slab(cur, cur + A_HALFS, wave, lrow, lhalf, c);
        if (more) sstore_slab(nxt, nxt + A_HALFS, tid, r);
        __syncthreads();
    }
#endif

    // C/D layout: VGPR i -> M = i + 8*lhalf, N = lane&15.
    float* py = y + (long)(mBase + wave * 16 + lhalf * 8) * OUT_F + n0 + lrow;
#pragma unroll
    for (int i = 0; i < 8; ++i) {
        py[(long)i * OUT_F +  0] = c[0][i];
        py[(long)i * OUT_F + 16] = c[1][i];
        py[(long)i * OUT_F + 32] = c[2][i];
        py[(long)i * OUT_F + 48] = c[3][i];
    }
}

// ---------------------------------------------------------------------------
// Fallback (workspace too small): plain VALU SpMM, exact fp32.
// ---------------------------------------------------------------------------
__global__ void spmm_fallback(const float* __restrict__ x,
                              const float* __restrict__ data,
                              const int*   __restrict__ indices,
                              const int*   __restrict__ indptr,
                              float*       __restrict__ y) {
    int r = blockIdx.x;
    int s = indptr[r], e = indptr[r + 1];
    for (int tok = threadIdx.x; tok < N_TOK; tok += blockDim.x) {
        float acc = 0.0f;
        const float* xr = x + (long)tok * IN_F;
        for (int j = s; j < e; ++j) acc += data[j] * xr[indices[j]];
        y[(long)tok * OUT_F + r] = acc;
    }
}

// ---------------------------------------------------------------------------
extern "C" void kernel_launch(void* const* d_in, const int* in_sizes, int n_in,
                              void* d_out, int out_size, void* d_ws, size_t ws_size,
                              hipStream_t stream) {
    const float* x       = (const float*)d_in[0];
    const float* data    = (const float*)d_in[1];
    const int*   indices = (const int*)d_in[2];
    const int*   indptr  = (const int*)d_in[3];
    float*       y       = (float*)d_out;

    const size_t wbytes = (size_t)OUT_F * IN_F * sizeof(_Float16); // 8 MB dense W
    const size_t xbytes = (size_t)N_TOK * IN_F * sizeof(_Float16); // 2 MB f16 x

    if (ws_size >= wbytes + xbytes) {
        _Float16* wd = (_Float16*)d_ws;
        _Float16* xh = (_Float16*)((char*)d_ws + wbytes);

        long n16 = (long)(wbytes / 16);
        zero_wd<<<(unsigned)((n16 + 255) / 256), 256, 0, stream>>>((uint4*)wd, n16);

        long nx = (long)N_TOK * IN_F;
        cvt_x<<<(unsigned)((nx + 255) / 256), 256, 0, stream>>>(x, xh, nx);

        scatter_csr<<<OUT_F / 256, 256, 0, stream>>>(data, indices, indptr, wd);

        dim3 grid(N_TOK / 128, OUT_F / 64);
        spmm_gemm_wmma<<<grid, 256, 0, stream>>>(xh, wd, y);
    } else {
        spmm_fallback<<<OUT_F, 256, 0, stream>>>(x, data, indices, indptr, y);
    }
}
